// FullyContextualEmbedding_63213328662731
// MI455X (gfx1250) — compile-verified, matrix operand order checked by reading, not run on hardware
//
#include <hip/hip_runtime.h>
#include <hip/hip_bf16.h>

// ---------------------------------------------------------------------------
// FullyContextualEmbedding on MI455X (gfx1250, wave32, WMMA).
//
// Rows are independent across the K=64 recurrent steps -> one persistent
// kernel, 32 rows per workgroup, all 64 steps inside with workgroup barriers.
// Precompute (once): base = f @ W_ih[:, :D].T + b_ih + b_hh   (step-invariant)
// Per step:          S = h@G^T -> softmax -> r = a@G
//                    gates = base + r@Wr^T + h@Whh^T ; LSTM update (+f residual)
// All GEMMs: v_wmma_f32_16x16x32_bf16 (f32 accumulate, bf16 operands).
// 32-row tiles double the FLOP/byte on the L2-resident weight stream vs 16.
// ---------------------------------------------------------------------------

typedef __bf16 bf16_t;
typedef __attribute__((ext_vector_type(16))) __bf16 v16bf;
typedef __attribute__((ext_vector_type(8)))  __bf16 v8bf;
typedef __attribute__((ext_vector_type(8)))  float  v8f;

#define NROWS   8192
#define DDIM    512
#define KSUP    64
#define G4D     2048   // 4*D
#define MTILE   32     // rows per workgroup (2 row-tiles of 16)

static __device__ __forceinline__ v8f wmma_bf16(v16bf a, v16bf b, v8f c) {
  return __builtin_amdgcn_wmma_f32_16x16x32_bf16(false, a, false, b, (short)0, c,
                                                 false, false);
}

// 16x32 bf16 A-fragment (or 32x16 B-fragment; symmetric per-lane layout) from
// a row-major bf16 buffer: two contiguous 16-byte loads per lane.
//  lane<16 : elems[0..7]=K(k0..k0+7)   elems[8..15]=K(k0+16..k0+23)
//  lane>=16: elems[0..7]=K(k0+8..+15)  elems[8..15]=K(k0+24..+31)
static __device__ __forceinline__ v16bf load_frag(const bf16_t* __restrict__ base,
                                                  int row_stride, int k0, int lane) {
  const int r  = lane & 15;
  const int hi = lane >> 4;
  const bf16_t* p = base + (size_t)r * row_stride + k0 + hi * 8;
  v8bf lo = *(const v8bf*)p;
  v8bf hh = *(const v8bf*)(p + 16);
  v16bf out;
#pragma unroll
  for (int i = 0; i < 8; ++i) { out[i] = lo[i]; out[i + 8] = hh[i]; }
  return out;
}

// C/D fragment element (v,lane) -> row = v + 8*(lane>>4), col = lane&15
static __device__ __forceinline__ void store_frag_f32(float* base, int row_stride,
                                                      v8f c, int lane) {
  const int col = lane & 15, hi = lane >> 4;
#pragma unroll
  for (int v = 0; v < 8; ++v)
    base[(size_t)(v + 8 * hi) * row_stride + col] = c[v];
}

static __device__ __forceinline__ void store_frag_bf16(bf16_t* base, int row_stride,
                                                       v8f c, int lane) {
  const int col = lane & 15, hi = lane >> 4;
#pragma unroll
  for (int v = 0; v < 8; ++v)
    base[(size_t)(v + 8 * hi) * row_stride + col] = (bf16_t)c[v];
}

static __device__ __forceinline__ float sigmoidf(float x) {
  return 1.0f / (1.0f + __expf(-x));
}
static __device__ __forceinline__ float fast_tanhf(float x) {
  return 2.0f / (1.0f + __expf(-2.0f * x)) - 1.0f;   // avoids ocml tanh expansion
}

// ---------------------------------------------------------------------------
// Precompute kernel 1: bf16 conversions / weight slicing.
// ---------------------------------------------------------------------------
__global__ void fce_convert(const float* __restrict__ f, const float* __restrict__ W_ih,
                            const float* __restrict__ W_hh,
                            bf16_t* __restrict__ f_b, bf16_t* __restrict__ Wf_b,
                            bf16_t* __restrict__ Wr_b, bf16_t* __restrict__ Whh_b) {
  const int stride = gridDim.x * blockDim.x;
  const int t = blockIdx.x * blockDim.x + threadIdx.x;
  for (int i = t; i < NROWS * DDIM; i += stride) f_b[i] = (bf16_t)f[i];
  for (int i = t; i < G4D * DDIM; i += stride) {
    const int j = i >> 9, d = i & (DDIM - 1);
    Wf_b[i]  = (bf16_t)W_ih[(size_t)j * (2 * DDIM) + d];
    Wr_b[i]  = (bf16_t)W_ih[(size_t)j * (2 * DDIM) + DDIM + d];
    Whh_b[i] = (bf16_t)W_hh[i];
  }
}

// ---------------------------------------------------------------------------
// Precompute kernel 2: base[n][j] = b_ih[j] + b_hh[j] + sum_d f[n][d]*W_ih[j][d]
// ---------------------------------------------------------------------------
__global__ void __launch_bounds__(256) fce_base(const bf16_t* __restrict__ f_b,
                                                const bf16_t* __restrict__ Wf_b,
                                                const float* __restrict__ b_ih,
                                                const float* __restrict__ b_hh,
                                                float* __restrict__ base) {
  const int lane = threadIdx.x & 31, wave = threadIdx.x >> 5;
  const int col = lane & 15, hi = lane >> 4;
  const int m0 = blockIdx.x * 16;
  const int j0 = blockIdx.y * 128 + wave * 16;

  const float bias = b_ih[j0 + col] + b_hh[j0 + col];
  v8f acc;
#pragma unroll
  for (int v = 0; v < 8; ++v) acc[v] = bias;

  for (int kc = 0; kc < DDIM / 32; ++kc) {
    v16bf A = load_frag(f_b + (size_t)m0 * DDIM, DDIM, kc * 32, lane);
    v16bf B = load_frag(Wf_b + (size_t)j0 * DDIM, DDIM, kc * 32, lane);
    acc = wmma_bf16(A, B, acc);
  }
#pragma unroll
  for (int v = 0; v < 8; ++v)
    base[(size_t)(m0 + v + 8 * hi) * G4D + j0 + col] = acc[v];
}

// ---------------------------------------------------------------------------
// Main persistent kernel: 256 blocks x 256 threads; each block owns 32 rows.
//
// LDS (~236 KB): G[64][512]bf16, Gt[512][64]bf16, h[2][32][512]bf16 (dbl buf),
//                r[32][512]bf16, S[32][64]f32, a[32][64]bf16
// ---------------------------------------------------------------------------
#define LDS_G    0
#define LDS_GT   (64 * 512 * 2)
#define LDS_H    (LDS_GT + 512 * 64 * 2)
#define LDS_R    (LDS_H + 2 * MTILE * 512 * 2)
#define LDS_S    (LDS_R + MTILE * 512 * 2)
#define LDS_A    (LDS_S + MTILE * 64 * 4)
#define LDS_TOT  (LDS_A + MTILE * 64 * 2)

__global__ void __launch_bounds__(256, 1) fce_main(const float* __restrict__ f,
                                                   const float* __restrict__ G,
                                                   const bf16_t* __restrict__ Wr,
                                                   const bf16_t* __restrict__ Whh,
                                                   const float* __restrict__ base,
                                                   float* __restrict__ out) {
  extern __shared__ char smem[];
  bf16_t* G_lds  = (bf16_t*)(smem + LDS_G);
  bf16_t* Gt_lds = (bf16_t*)(smem + LDS_GT);
  bf16_t* h_lds  = (bf16_t*)(smem + LDS_H);
  bf16_t* r_lds  = (bf16_t*)(smem + LDS_R);
  float*  S_lds  = (float*)(smem + LDS_S);
  bf16_t* a_lds  = (bf16_t*)(smem + LDS_A);

  const int tid = threadIdx.x;
  const int lane = tid & 31, wave = tid >> 5;
  const int col = lane & 15, hi = lane >> 4;
  const int row0 = blockIdx.x * MTILE;

  // Stage G in both layouts (bf16) once.
  for (int i = tid; i < KSUP * DDIM; i += 256) {
    const float g = G[i];
    const int r = i >> 9, c = i & (DDIM - 1);
    G_lds[i] = (bf16_t)g;
    Gt_lds[c * KSUP + r] = (bf16_t)g;
  }

  // Cell state in registers: 2 row-tiles x 4 d-tiles {wave, wave+8, +16, +24}.
  v8f c_frag[2][4];
#pragma unroll
  for (int rt = 0; rt < 2; ++rt)
#pragma unroll
    for (int g = 0; g < 4; ++g)
#pragma unroll
      for (int v = 0; v < 8; ++v) c_frag[rt][g][v] = 0.0f;

  // h0 = f (bf16) into buffer 0.
  for (int i = tid; i < MTILE * DDIM; i += 256)
    h_lds[i] = (bf16_t)f[(size_t)(row0 + (i >> 9)) * DDIM + (i & (DDIM - 1))];
  __syncthreads();

  int cur = 0;
  for (int step = 0; step < KSUP; ++step) {
    const bf16_t* hb = h_lds + cur * MTILE * DDIM;
    bf16_t* hb_next  = h_lds + (cur ^ 1) * MTILE * DDIM;

    // ---- S = h @ G^T  (32x64 = 8 tiles; one per wave) ----
    {
      const int rt = wave >> 2, kt = wave & 3;
      v8f s = {};
      for (int kc = 0; kc < DDIM / 32; ++kc) {
        v16bf A = load_frag(hb + (size_t)rt * 16 * DDIM, DDIM, kc * 32, lane);
        v16bf B = load_frag(G_lds + (size_t)kt * 16 * DDIM, DDIM, kc * 32, lane);
        s = wmma_bf16(A, B, s);
      }
      store_frag_f32(S_lds + rt * 16 * KSUP + kt * 16, KSUP, s, lane);
    }
    __syncthreads();

    // ---- row softmax over 64 scores (one full wave) ----
    if (tid < MTILE) {
      float m = -3.402823466e38f;
      for (int k = 0; k < KSUP; ++k) m = fmaxf(m, S_lds[tid * KSUP + k]);
      float sum = 0.0f;
      for (int k = 0; k < KSUP; ++k) sum += __expf(S_lds[tid * KSUP + k] - m);
      const float inv = 1.0f / sum;
      for (int k = 0; k < KSUP; ++k)
        a_lds[tid * KSUP + k] = (bf16_t)(__expf(S_lds[tid * KSUP + k] - m) * inv);
    }
    __syncthreads();

    // ---- r = a @ G  (32x512; 2 row-tiles x 4 d-tiles per wave) ----
#pragma unroll
    for (int g = 0; g < 4; ++g) {
      const int dt = wave + 8 * g;
#pragma unroll
      for (int rt = 0; rt < 2; ++rt) {
        v8f racc = {};
#pragma unroll
        for (int kc = 0; kc < KSUP / 32; ++kc) {
          v16bf A = load_frag(a_lds + (size_t)rt * 16 * KSUP, KSUP, kc * 32, lane);
          v16bf B = load_frag(Gt_lds + (size_t)dt * 16 * KSUP, KSUP, kc * 32, lane);
          racc = wmma_bf16(A, B, racc);
        }
        store_frag_bf16(r_lds + (size_t)rt * 16 * DDIM + dt * 16, DDIM, racc, lane);
      }
    }
    __syncthreads();

    // ---- gates = base + r@Wr^T + h@Whh^T ; fused LSTM update ----
    // Each weight B-fragment feeds 2 WMMAs (both row-tiles): 16 WMMAs per
    // k-chunk against 16 global b128 + 8 ds b128 loads.
#pragma unroll
    for (int g = 0; g < 4; ++g) {
      const int dt = wave + 8 * g;
      v8f acc[2][4];
#pragma unroll
      for (int rt = 0; rt < 2; ++rt)
#pragma unroll
        for (int q = 0; q < 4; ++q) {
          const int jt = q * 32 + dt;
#pragma unroll
          for (int v = 0; v < 8; ++v)
            acc[rt][q][v] =
                base[(size_t)(row0 + rt * 16 + v + 8 * hi) * G4D + jt * 16 + col];
        }
      for (int kc = 0; kc < DDIM / 32; ++kc) {
        v16bf Ah0 = load_frag(hb, DDIM, kc * 32, lane);
        v16bf Ah1 = load_frag(hb + 16 * DDIM, DDIM, kc * 32, lane);
        v16bf Ar0 = load_frag(r_lds, DDIM, kc * 32, lane);
        v16bf Ar1 = load_frag(r_lds + 16 * DDIM, DDIM, kc * 32, lane);
#pragma unroll
        for (int q = 0; q < 4; ++q) {
          const int jt = q * 32 + dt;
          v16bf Bh = load_frag(Whh + (size_t)jt * 16 * DDIM, DDIM, kc * 32, lane);
          acc[0][q] = wmma_bf16(Ah0, Bh, acc[0][q]);
          acc[1][q] = wmma_bf16(Ah1, Bh, acc[1][q]);
          v16bf Br = load_frag(Wr + (size_t)jt * 16 * DDIM, DDIM, kc * 32, lane);
          acc[0][q] = wmma_bf16(Ar0, Br, acc[0][q]);
          acc[1][q] = wmma_bf16(Ar1, Br, acc[1][q]);
        }
      }
#pragma unroll
      for (int rt = 0; rt < 2; ++rt)
#pragma unroll
        for (int v = 0; v < 8; ++v) {
          const size_t row = (size_t)(row0 + rt * 16 + v + 8 * hi);
          const int colg = dt * 16 + col;
          const float fv = f[row * DDIM + colg];           // L2-resident reload
          const float ig = sigmoidf(acc[rt][0][v]);
          const float fg = sigmoidf(acc[rt][1][v]);
          const float gg = fast_tanhf(acc[rt][2][v]);
          const float og = sigmoidf(acc[rt][3][v]);
          const float cn = fg * c_frag[rt][g][v] + ig * gg;
          const float hn = og * fast_tanhf(cn) + fv;       // residual +f
          c_frag[rt][g][v] = cn;
          hb_next[(size_t)(rt * 16 + v + 8 * hi) * DDIM + colg] = (bf16_t)hn;
          if (step == KSUP - 1) out[row * DDIM + colg] = hn;
        }
    }
    cur ^= 1;
    __syncthreads();
  }
}

// ---------------------------------------------------------------------------
extern "C" void kernel_launch(void* const* d_in, const int* in_sizes, int n_in,
                              void* d_out, int out_size, void* d_ws, size_t ws_size,
                              hipStream_t stream) {
  const float* f    = (const float*)d_in[0];
  const float* G    = (const float*)d_in[1];
  const float* W_ih = (const float*)d_in[2];
  const float* W_hh = (const float*)d_in[3];
  const float* b_ih = (const float*)d_in[4];
  const float* b_hh = (const float*)d_in[5];
  float* out = (float*)d_out;

  // Workspace carve-up (~78 MB):
  char* ws = (char*)d_ws;
  bf16_t* f_b   = (bf16_t*)(ws);                          //  8 MB
  bf16_t* Wf_b  = (bf16_t*)(ws + (size_t)(8)  * 1048576); //  2 MB
  bf16_t* Wr_b  = (bf16_t*)(ws + (size_t)(10) * 1048576); //  2 MB
  bf16_t* Whh_b = (bf16_t*)(ws + (size_t)(12) * 1048576); //  2 MB
  float*  base  = (float*)(ws + (size_t)(14) * 1048576);  // 64 MB (L2-resident)

  fce_convert<<<1024, 256, 0, stream>>>(f, W_ih, W_hh, f_b, Wf_b, Wr_b, Whh_b);
  fce_base<<<dim3(NROWS / 16, G4D / 128), 256, 0, stream>>>(f_b, Wf_b, b_ih, b_hh, base);

  static_assert(LDS_TOT <= 320 * 1024, "LDS budget");
  hipFuncSetAttribute((const void*)fce_main,
                      hipFuncAttributeMaxDynamicSharedMemorySize, LDS_TOT);
  fce_main<<<NROWS / MTILE, 256, LDS_TOT, stream>>>(f, G, Wr_b, Whh_b, base, out);
}
